// RGCN_71743133712990
// MI455X (gfx1250) — compile-verified, matrix operand order checked by reading, not run on hardware
//
#include <hip/hip_runtime.h>
#include <hip/hip_bf16.h>
#include <stdint.h>

// ---------------------------------------------------------------------------
// RGCN forward (basis decomposition) for MI455X / gfx1250, wave32 + WMMA bf16
// ---------------------------------------------------------------------------

typedef __attribute__((ext_vector_type(16))) __bf16 v16bf;
typedef __attribute__((ext_vector_type(4)))  __bf16 v4bf;
typedef __attribute__((ext_vector_type(8)))  float  v8f;

#define N_ENT  30000
#define R_EFF  9
#define F_IN   256
#define HID    256
#define NCLS   32
#define BASES  16
#define NE     500000
#define RN     (R_EFF * N_ENT)         // 270000 aggregated rows
#define M_TILES (N_ENT / 16)           // 1875 (exact)

union FragAB {                          // 16 bf16 = two 16-byte chunks
    uint4 u[2];
    v16bf v;
};

// ---------------------------------------------------------------- utilities
__global__ void zero_f32_kernel(float* __restrict__ p, int n4) {
    int i = blockIdx.x * blockDim.x + threadIdx.x;
    if (i < n4) reinterpret_cast<float4*>(p)[i] = make_float4(0.f, 0.f, 0.f, 0.f);
}

__global__ void f32_to_bf16_kernel(const float* __restrict__ src,
                                   __bf16* __restrict__ dst, int n4) {
    int i = blockIdx.x * blockDim.x + threadIdx.x;
    if (i >= n4) return;
    float4 f = reinterpret_cast<const float4*>(src)[i];
    v4bf o;
    o.x = (__bf16)f.x; o.y = (__bf16)f.y; o.z = (__bf16)f.z; o.w = (__bf16)f.w;
    reinterpret_cast<v4bf*>(dst)[i] = o;
}

// ------------------------------------------------- W = comps @ bases, N-major
// w1t[r][e][f] = sum_b comps1[r,b] * bases1[b,f,e]   (bf16, K-contiguous)
// w2t[r][c][e] = sum_b comps2[r,b] * bases2[b,e,c]
__global__ void prep_weights_kernel(const float* __restrict__ comps1,
                                    const float* __restrict__ bases1,
                                    const float* __restrict__ comps2,
                                    const float* __restrict__ bases2,
                                    __bf16* __restrict__ w1t,
                                    __bf16* __restrict__ w2t) {
    int idx = blockIdx.x * blockDim.x + threadIdx.x;
    const int n1 = R_EFF * HID * F_IN;
    const int n2 = R_EFF * NCLS * HID;
    if (idx < n1) {
        int r = idx / (HID * F_IN);
        int e = (idx / F_IN) % HID;
        int f = idx % F_IN;
        float s = 0.f;
#pragma unroll
        for (int b = 0; b < BASES; ++b)
            s += comps1[r * BASES + b] * bases1[((size_t)b * F_IN + f) * HID + e];
        w1t[idx] = (__bf16)s;
    } else if (idx < n1 + n2) {
        int j = idx - n1;
        int r = j / (NCLS * HID);
        int c = (j / HID) % NCLS;
        int e = j % HID;
        float s = 0.f;
#pragma unroll
        for (int b = 0; b < BASES; ++b)
            s += comps2[r * BASES + b] * bases2[((size_t)b * HID + e) * NCLS + c];
        w2t[j] = (__bf16)s;
    }
}

// ------------------------------------------------------------- SpMM (atomics)
// hf[row, :] += val * x[col, :]  over E edges; 4 floats per thread.
__global__ void spmm_f32_kernel(const float* __restrict__ x,
                                const float* __restrict__ vals,
                                const int* __restrict__ rows,
                                const int* __restrict__ cols,
                                float* __restrict__ hf) {
    int idx = blockIdx.x * blockDim.x + threadIdx.x;
    if (idx >= NE * 64) return;
    int e = idx >> 6;
    int j = (idx & 63) << 2;
    int row = rows[e];
    int col = cols[e];
    float v = vals[e];
    float4 xv = *reinterpret_cast<const float4*>(x + (size_t)col * F_IN + j);
    float* dst = hf + (size_t)row * F_IN + j;
    __hip_atomic_fetch_add(dst + 0, v * xv.x, __ATOMIC_RELAXED, __HIP_MEMORY_SCOPE_AGENT);
    __hip_atomic_fetch_add(dst + 1, v * xv.y, __ATOMIC_RELAXED, __HIP_MEMORY_SCOPE_AGENT);
    __hip_atomic_fetch_add(dst + 2, v * xv.z, __ATOMIC_RELAXED, __HIP_MEMORY_SCOPE_AGENT);
    __hip_atomic_fetch_add(dst + 3, v * xv.w, __ATOMIC_RELAXED, __HIP_MEMORY_SCOPE_AGENT);
}

// Layer-2 variant: gather from bf16 hidden activations.
__global__ void spmm_bf16_kernel(const __bf16* __restrict__ x,
                                 const float* __restrict__ vals,
                                 const int* __restrict__ rows,
                                 const int* __restrict__ cols,
                                 float* __restrict__ hf) {
    int idx = blockIdx.x * blockDim.x + threadIdx.x;
    if (idx >= NE * 64) return;
    int e = idx >> 6;
    int j = (idx & 63) << 2;
    int row = rows[e];
    int col = cols[e];
    float v = vals[e];
    union { uint2 u; __bf16 h[4]; } xv;
    xv.u = *reinterpret_cast<const uint2*>(x + (size_t)col * HID + j);
    float* dst = hf + (size_t)row * HID + j;
    __hip_atomic_fetch_add(dst + 0, v * (float)xv.h[0], __ATOMIC_RELAXED, __HIP_MEMORY_SCOPE_AGENT);
    __hip_atomic_fetch_add(dst + 1, v * (float)xv.h[1], __ATOMIC_RELAXED, __HIP_MEMORY_SCOPE_AGENT);
    __hip_atomic_fetch_add(dst + 2, v * (float)xv.h[2], __ATOMIC_RELAXED, __HIP_MEMORY_SCOPE_AGENT);
    __hip_atomic_fetch_add(dst + 3, v * (float)xv.h[3], __ATOMIC_RELAXED, __HIP_MEMORY_SCOPE_AGENT);
}

// ----------------------------------------------------------- GEMM1 via WMMA
// out1[n, e] = relu( sum_r sum_f hb[r*N+n, f] * w1t[r, e, f] + bias1[e] )
// Wave: one 16-row M-tile x four 16-col N-tiles (A fragment reused 4x).
__global__ void __launch_bounds__(256)
gemm1_wmma_kernel(const __bf16* __restrict__ hb,
                  const __bf16* __restrict__ w1t,
                  const float* __restrict__ bias1,
                  __bf16* __restrict__ out1b) {
    const int lane  = threadIdx.x & 31;
    const int wave  = threadIdx.x >> 5;
    const int mrow  = lane & 15;        // A: row within tile, B: col within tile
    const int half  = lane >> 4;
    const int kbase = half * 8;         // ISA 16-bit A 16x32 lane layout
    const int m_tile = blockIdx.x * 2 + (wave >> 2);
    const int nb     = (wave & 3) * 64; // 4 waves cover N = 256
    if (m_tile >= M_TILES) return;
    const int row0 = m_tile * 16;

    v8f acc[4] = {};
    for (int r = 0; r < R_EFF; ++r) {
        const __bf16* Ar = hb  + ((size_t)(r * N_ENT + row0 + mrow)) * F_IN;
        const __bf16* Br = w1t + ((size_t)(r * HID  + nb   + mrow)) * F_IN;
#pragma unroll 2
        for (int k0 = 0; k0 < F_IN; k0 += 32) {
            FragAB a;
            a.u[0] = *reinterpret_cast<const uint4*>(Ar + k0 + kbase);
            a.u[1] = *reinterpret_cast<const uint4*>(Ar + k0 + kbase + 16);
#pragma unroll
            for (int t = 0; t < 4; ++t) {
                const __bf16* Bp = Br + (size_t)t * 16 * F_IN + k0 + kbase;
                FragAB b;
                b.u[0] = *reinterpret_cast<const uint4*>(Bp);
                b.u[1] = *reinterpret_cast<const uint4*>(Bp + 16);
                acc[t] = __builtin_amdgcn_wmma_f32_16x16x32_bf16(
                    false, a.v, false, b.v, (short)0, acc[t], false, false);
            }
        }
    }
    // Epilogue: +bias, ReLU, store bf16. C/D layout: lane covers col = mrow,
    // VGPR g -> row g + 8*half.
#pragma unroll
    for (int t = 0; t < 4; ++t) {
        const int col = nb + t * 16 + mrow;
        const float bv = bias1[col];
#pragma unroll
        for (int g = 0; g < 8; ++g) {
            float v = acc[t][g] + bv;
            v = fmaxf(v, 0.f);
            const int row = row0 + g + half * 8;
            out1b[(size_t)row * HID + col] = (__bf16)v;
        }
    }
}

// ----------------------------------------------------------- GEMM2 via WMMA
// out[n, c] = sum_r sum_e hb[r*N+n, e] * w2t[r, c, e] + bias2[c]
// Wave: one 16-row M-tile x two 16-col N-tiles (N = 32 total).
__global__ void __launch_bounds__(256)
gemm2_wmma_kernel(const __bf16* __restrict__ hb,
                  const __bf16* __restrict__ w2t,
                  const float* __restrict__ bias2,
                  float* __restrict__ out) {
    const int lane  = threadIdx.x & 31;
    const int wave  = threadIdx.x >> 5;
    const int mrow  = lane & 15;
    const int half  = lane >> 4;
    const int kbase = half * 8;
    const int m_tile = blockIdx.x * 8 + wave;
    if (m_tile >= M_TILES) return;
    const int row0 = m_tile * 16;

    v8f acc[2] = {};
    for (int r = 0; r < R_EFF; ++r) {
        const __bf16* Ar = hb  + ((size_t)(r * N_ENT + row0 + mrow)) * HID;
        const __bf16* Br = w2t + ((size_t)(r * NCLS  + mrow)) * HID;
#pragma unroll 2
        for (int k0 = 0; k0 < HID; k0 += 32) {
            FragAB a;
            a.u[0] = *reinterpret_cast<const uint4*>(Ar + k0 + kbase);
            a.u[1] = *reinterpret_cast<const uint4*>(Ar + k0 + kbase + 16);
#pragma unroll
            for (int t = 0; t < 2; ++t) {
                const __bf16* Bp = Br + (size_t)t * 16 * HID + k0 + kbase;
                FragAB b;
                b.u[0] = *reinterpret_cast<const uint4*>(Bp);
                b.u[1] = *reinterpret_cast<const uint4*>(Bp + 16);
                acc[t] = __builtin_amdgcn_wmma_f32_16x16x32_bf16(
                    false, a.v, false, b.v, (short)0, acc[t], false, false);
            }
        }
    }
#pragma unroll
    for (int t = 0; t < 2; ++t) {
        const int col = t * 16 + mrow;
        const float bv = bias2[col];
#pragma unroll
        for (int g = 0; g < 8; ++g) {
            const int row = row0 + g + half * 8;
            out[(size_t)row * NCLS + col] = acc[t][g] + bv;
        }
    }
}

// ---------------------------------------------------------------------------
extern "C" void kernel_launch(void* const* d_in, const int* in_sizes, int n_in,
                              void* d_out, int out_size, void* d_ws, size_t ws_size,
                              hipStream_t stream) {
    const float* features = (const float*)d_in[0];
    const float* ver_vals = (const float*)d_in[1];
    const float* comps1   = (const float*)d_in[2];
    const float* bases1   = (const float*)d_in[3];
    const float* comps2   = (const float*)d_in[4];
    const float* bases2   = (const float*)d_in[5];
    const float* bias1    = (const float*)d_in[6];
    const float* bias2    = (const float*)d_in[7];
    const int*   ver_rows = (const int*)d_in[8];
    const int*   ver_cols = (const int*)d_in[9];

    // Workspace carve-up (256B aligned). hf/hb are reused by both layers.
    char*  ws  = (char*)d_ws;
    size_t off = 0;
    auto carve = [&](size_t bytes) -> void* {
        void* p = ws + off;
        off = (off + bytes + 255) & ~(size_t)255;
        return p;
    };
    float*  hf    = (float*) carve((size_t)RN * 256 * sizeof(float));   // 276 MB
    __bf16* hb    = (__bf16*)carve((size_t)RN * 256 * sizeof(__bf16));  // 138 MB
    __bf16* out1b = (__bf16*)carve((size_t)N_ENT * HID * sizeof(__bf16));
    __bf16* w1t   = (__bf16*)carve((size_t)R_EFF * HID * F_IN * sizeof(__bf16));
    __bf16* w2t   = (__bf16*)carve((size_t)R_EFF * NCLS * HID * sizeof(__bf16));

    const int T = 256;
    const int n_prep = R_EFF * HID * F_IN + R_EFF * NCLS * HID;   // 663552
    const int n4h    = RN * 256 / 4;                               // 17.28M float4
    const int n_spmm = NE * 64;                                    // 32M threads

    // 0) fused basis weights, transposed to K-contiguous layout
    prep_weights_kernel<<<(n_prep + T - 1) / T, T, 0, stream>>>(
        comps1, bases1, comps2, bases2, w1t, w2t);

    // ----- Layer 1 -----
    zero_f32_kernel<<<(n4h + T - 1) / T, T, 0, stream>>>(hf, n4h);
    spmm_f32_kernel<<<(n_spmm + T - 1) / T, T, 0, stream>>>(
        features, ver_vals, ver_rows, ver_cols, hf);
    f32_to_bf16_kernel<<<(n4h + T - 1) / T, T, 0, stream>>>(hf, hb, n4h);
    gemm1_wmma_kernel<<<(M_TILES + 1) / 2, T, 0, stream>>>(hb, w1t, bias1, out1b);

    // ----- Layer 2 -----
    zero_f32_kernel<<<(n4h + T - 1) / T, T, 0, stream>>>(hf, n4h);
    spmm_bf16_kernel<<<(n_spmm + T - 1) / T, T, 0, stream>>>(
        out1b, ver_vals, ver_rows, ver_cols, hf);
    f32_to_bf16_kernel<<<(n4h + T - 1) / T, T, 0, stream>>>(hf, hb, n4h);
    gemm2_wmma_kernel<<<(M_TILES + 7) / 8, T, 0, stream>>>(
        hb, w2t, bias2, (float*)d_out);
}
